// MemoryEfficientAttention_70609262346600
// MI455X (gfx1250) — matrix operand order repaired and checked by reading, not compiled, not run on hardware
//
#include <hip/hip_runtime.h>
#include <hip/hip_bf16.h>

typedef unsigned short u16;
typedef __attribute__((ext_vector_type(16))) __bf16 bf16x16;
typedef __attribute__((ext_vector_type(8)))  float  f32x8;
typedef int v4i __attribute__((ext_vector_type(4)));
typedef __attribute__((address_space(1))) v4i gas_v4i;   // global
typedef __attribute__((address_space(3))) v4i las_v4i;   // LDS

#define Bb   2
#define Ss   2048
#define Dd   2048
#define Hh   16
#define HDd  128
#define CHK  1024
#define Mm   (Bb*Ss)

#define EPI_QKV        1
#define EPI_RESID      2
#define EPI_BIAS_GELU  3
#define EPI_BIAS_RESID 4

#if defined(__has_builtin)
#  if __has_builtin(__builtin_amdgcn_global_load_async_to_lds_b128) && \
      __has_builtin(__builtin_amdgcn_s_wait_asynccnt)
#    define USE_ASYNC 1
#  endif
#endif
#ifndef USE_ASYNC
#  define USE_ASYNC 0
#endif

__device__ __forceinline__ u16 f32_to_bf16(float f) {
  unsigned u = __float_as_uint(f);
  u += 0x7fffu + ((u >> 16) & 1u);            // round-to-nearest-even
  return (u16)(u >> 16);
}

// A-fragment (16x32, lane = M row, K in groups of 8 per half-wave): two b128 loads
__device__ __forceinline__ bf16x16 load_a_frag(const u16* p0, int stride, int lane) {
  const u16* p = p0 + (size_t)(lane & 15) * stride + ((lane >> 4) << 3);
  union { bf16x16 v; uint4 q[2]; } u;
  u.q[0] = *(const uint4*)(p);
  u.q[1] = *(const uint4*)(p + 16);
  return u.v;
}
// B-fragment (32x16, lane = N col, K in groups of 8 per half-wave): two b128 loads
__device__ __forceinline__ bf16x16 load_b_frag(const u16* p0, int stride, int lane) {
  const u16* p = p0 + (size_t)(lane & 15) * stride + ((lane >> 4) << 3);
  union { bf16x16 v; uint4 q[2]; } u;
  u.q[0] = *(const uint4*)(p);
  u.q[1] = *(const uint4*)(p + 16);
  return u.v;
}

__device__ __forceinline__ f32x8 wmma_bf16(bf16x16 a, bf16x16 b, f32x8 c) {
  return __builtin_amdgcn_wmma_f32_16x16x32_bf16(false, a, false, b, (short)0, c, false, false);
}

// ---------------- weight transpose + cast: w[K][N] f32 -> wt[N][K] bf16 ----------------
__global__ void transpose_cast_kernel(const float* __restrict__ w, u16* __restrict__ wt,
                                      int K, int N) {
  int idx = blockIdx.x * blockDim.x + threadIdx.x;
  if (idx >= K * N) return;
  int n = idx / K, k = idx % K;
  wt[idx] = f32_to_bf16(w[(size_t)k * N + n]);
}

// ---------------- LayerNorm (f32 in, bf16 out), one 256-thread block per row ----------------
__global__ __launch_bounds__(256) void layernorm_kernel(const float* __restrict__ x,
                                                        const float* __restrict__ g,
                                                        const float* __restrict__ be,
                                                        u16* __restrict__ out, int N) {
  int row = blockIdx.x;
  const float* xr = x + (size_t)row * N;
  float s = 0.f, s2 = 0.f;
  for (int c = threadIdx.x; c < N; c += 256) { float v = xr[c]; s += v; s2 += v * v; }
  __shared__ float red0[8], red1[8];
  for (int off = 16; off; off >>= 1) { s += __shfl_xor(s, off, 32); s2 += __shfl_xor(s2, off, 32); }
  int wid = threadIdx.x >> 5, ln = threadIdx.x & 31;
  if (ln == 0) { red0[wid] = s; red1[wid] = s2; }
  __syncthreads();
  if (wid == 0) {
    s = (ln < 8) ? red0[ln] : 0.f; s2 = (ln < 8) ? red1[ln] : 0.f;
    for (int off = 4; off; off >>= 1) { s += __shfl_xor(s, off, 32); s2 += __shfl_xor(s2, off, 32); }
    if (ln == 0) { red0[0] = s; red1[0] = s2; }
  }
  __syncthreads();
  float mean = red0[0] / N;
  float var  = red1[0] / N - mean * mean;
  float inv  = rsqrtf(var + 1e-5f);
  u16* orow = out + (size_t)row * N;
  for (int c = threadIdx.x; c < N; c += 256)
    orow[c] = f32_to_bf16((xr[c] - mean) * inv * g[c] + be[c]);
}

// ---------------- bf16 WMMA GEMM, double-buffered LDS, async global->LDS copies ----------------
// C[M,N] = A[M,K] @ BT[N,K]^T with fused epilogues.
template <int EPI>
__global__ __launch_bounds__(256) void gemm_kernel(
    const u16* __restrict__ A, const u16* __restrict__ BT,
    int Mdim, int Ndim, int Kdim,
    const float* __restrict__ bias, const float* __restrict__ resid,
    float* __restrict__ outf, u16* __restrict__ outb,
    u16* __restrict__ qo, u16* __restrict__ ko, u16* __restrict__ vo) {
  __shared__ alignas(16) u16 As[2][128 * 40];
  __shared__ alignas(16) u16 Bs[2][128 * 40];
  int tid = threadIdx.x;
  int lane = tid & 31, wave = tid >> 5;
  int wm = wave & 1, wn = wave >> 1;        // 2x4 wave grid -> 64x32 per wave
  int bm = blockIdx.y * 128, bn = blockIdx.x * 128;

  // Each thread moves 2 A-chunks + 2 B-chunks of 16B per K-tile.
  auto issue_tile = [&](int buf, int k0) {
#pragma unroll
    for (int ch = tid; ch < 512; ch += 256) {
      int r = ch >> 2, cc = ch & 3;
      const u16* ga = &A [(size_t)(bm + r) * Kdim + k0 + cc * 8];
      const u16* gb = &BT[(size_t)(bn + r) * Kdim + k0 + cc * 8];
      u16* la = &As[buf][r * 40 + cc * 8];
      u16* lb = &Bs[buf][r * 40 + cc * 8];
#if USE_ASYNC
      __builtin_amdgcn_global_load_async_to_lds_b128((gas_v4i*)ga, (las_v4i*)la, 0, 0);
      __builtin_amdgcn_global_load_async_to_lds_b128((gas_v4i*)gb, (las_v4i*)lb, 0, 0);
#else
      *(uint4*)la = *(const uint4*)ga;
      *(uint4*)lb = *(const uint4*)gb;
      __builtin_prefetch(ga + 32);
      __builtin_prefetch(gb + 32);
#endif
    }
  };

  f32x8 zero8 = {0.f, 0.f, 0.f, 0.f, 0.f, 0.f, 0.f, 0.f};
  f32x8 acc[4][2];
#pragma unroll
  for (int i = 0; i < 4; i++)
#pragma unroll
    for (int j = 0; j < 2; j++) acc[i][j] = zero8;

  int nk = Kdim >> 5;
  issue_tile(0, 0);
  for (int t = 0; t < nk; ++t) {
#if USE_ASYNC
    __builtin_amdgcn_s_wait_asynccnt(0);
#endif
    __syncthreads();                       // tile t visible to all waves
    if (t + 1 < nk) issue_tile((t + 1) & 1, (t + 1) << 5);
    int cur = t & 1;
    bf16x16 af[4], bf[2];
#pragma unroll
    for (int i = 0; i < 4; i++) af[i] = load_a_frag(&As[cur][(wm * 64 + i * 16) * 40], 40, lane);
#pragma unroll
    for (int j = 0; j < 2; j++) bf[j] = load_b_frag(&Bs[cur][(wn * 32 + j * 16) * 40], 40, lane);
#pragma unroll
    for (int i = 0; i < 4; i++)
#pragma unroll
      for (int j = 0; j < 2; j++) acc[i][j] = wmma_bf16(af[i], bf[j], acc[i][j]);
  }

  int laneHalf = lane >> 4, coln = lane & 15;
#pragma unroll
  for (int i = 0; i < 4; i++)
#pragma unroll
    for (int j = 0; j < 2; j++)
#pragma unroll
      for (int r = 0; r < 8; r++) {
        int gm = bm + wm * 64 + i * 16 + r + 8 * laneHalf;
        int gn = bn + wn * 32 + j * 16 + coln;
        float v = acc[i][j][r];
        if (EPI == EPI_QKV) {
          int t = gn >> 11, rem = gn & 2047, hh = rem >> 7, d = rem & 127;
          int b = gm >> 11, s = gm & 2047;
          if (t == 0)
            qo[(((size_t)b * Hh + hh) * Ss + s) * HDd + d] = f32_to_bf16(v * 0.08838834764831845f);
          else if (t == 1)
            ko[(((size_t)b * Hh + hh) * Ss + s) * HDd + d] = f32_to_bf16(v);
          else
            vo[(((size_t)b * Hh + hh) * HDd + d) * Ss + s] = f32_to_bf16(v);
        } else if (EPI == EPI_RESID) {
          outf[(size_t)gm * Ndim + gn] = resid[(size_t)gm * Ndim + gn] + v;
        } else if (EPI == EPI_BIAS_GELU) {
          float xg = v + bias[gn];
          outb[(size_t)gm * Ndim + gn] =
              f32_to_bf16(0.5f * xg * (1.0f + erff(xg * 0.70710678118654752f)));
        } else {  // EPI_BIAS_RESID
          outf[(size_t)gm * Ndim + gn] = resid[(size_t)gm * Ndim + gn] + v + bias[gn];
        }
      }
}

// ---------------- flash-style attention: one wave per 16-query tile ----------------
__global__ __launch_bounds__(128) void attn_kernel(const u16* __restrict__ q,
                                                   const u16* __restrict__ k,
                                                   const u16* __restrict__ vT,
                                                   u16* __restrict__ attn_out) {
  int bh = blockIdx.x;                 // B*H = 32
  int b = bh >> 4, h = bh & 15;
  int wave = threadIdx.x >> 5;
  int lane = threadIdx.x & 31;
  int qt = blockIdx.y * 4 + wave;      // 128 q-tiles per (b,h)
  int q0 = qt * 16;
  int rloc0 = q0 & (CHK - 1);          // q % CHUNK for row 0

  const u16* qbase = q  + ((size_t)bh * Ss + q0) * HDd;
  const u16* kbp   = k  + (size_t)bh * Ss * HDd;
  const u16* vbase = vT + (size_t)bh * HDd * Ss;

  __shared__ alignas(16) u16 Pl[4][16 * 40];
  u16* P = &Pl[wave][0];

  bf16x16 qf[4];
#pragma unroll
  for (int c = 0; c < 4; c++) qf[c] = load_a_frag(qbase + c * 32, HDd, lane);

  f32x8 zero8 = {0.f, 0.f, 0.f, 0.f, 0.f, 0.f, 0.f, 0.f};
  f32x8 acc[8];
#pragma unroll
  for (int t = 0; t < 8; t++) acc[t] = zero8;
  float mrun[8], lrun[8];
#pragma unroll
  for (int r = 0; r < 8; r++) { mrun[r] = -1e30f; lrun[r] = 0.f; }

  int laneHalf = lane >> 4, coln = lane & 15;
  int kmax = rloc0 + 15;               // max allowed key index across this q tile

  for (int kb = 0; kb <= kmax; kb += 32) {
    f32x8 s0 = zero8, s1 = zero8;
#pragma unroll
    for (int c = 0; c < 4; c++) {
      bf16x16 k0f = load_b_frag(kbp + (size_t)kb * HDd + c * 32, HDd, lane);
      bf16x16 k1f = load_b_frag(kbp + (size_t)(kb + 16) * HDd + c * 32, HDd, lane);
      s0 = wmma_bf16(qf[c], k0f, s0);
      s1 = wmma_bf16(qf[c], k1f, s1);
    }
#pragma unroll
    for (int r = 0; r < 8; r++) {
      int row = r + 8 * laneHalf;
      int qmod = rloc0 + row;
      float v0 = s0[r], v1 = s1[r];
      if (kb + coln > qmod)      v0 = -1e9f;
      if (kb + 16 + coln > qmod) v1 = -1e9f;
      float mx = fmaxf(v0, v1);
      for (int off = 8; off; off >>= 1) mx = fmaxf(mx, __shfl_xor(mx, off, 32));
      float mnew = fmaxf(mrun[r], mx);
      float sc = __expf(mrun[r] - mnew);
      float p0 = __expf(v0 - mnew);
      float p1 = __expf(v1 - mnew);
      float ls = p0 + p1;
      for (int off = 8; off; off >>= 1) ls += __shfl_xor(ls, off, 32);
      lrun[r] = lrun[r] * sc + ls;
      mrun[r] = mnew;
#pragma unroll
      for (int t = 0; t < 8; t++) acc[t][r] *= sc;
      P[row * 40 + coln]      = f32_to_bf16(p0);
      P[row * 40 + coln + 16] = f32_to_bf16(p1);
    }
    asm volatile("s_wait_dscnt 0" ::: "memory");
    bf16x16 pf = load_a_frag(P, 40, lane);
#pragma unroll
    for (int t = 0; t < 8; t++) {
      bf16x16 vf = load_b_frag(vbase + (size_t)(t * 16) * Ss + kb, Ss, lane);
      acc[t] = wmma_bf16(pf, vf, acc[t]);
    }
    asm volatile("s_wait_dscnt 0" ::: "memory");
  }

#pragma unroll
  for (int r = 0; r < 8; r++) {
    float inv = 1.0f / lrun[r];
    int row = r + 8 * laneHalf;
    size_t base = ((size_t)b * Ss + (q0 + row)) * Dd + h * HDd;
#pragma unroll
    for (int t = 0; t < 8; t++)
      attn_out[base + t * 16 + coln] = f32_to_bf16(acc[t][r] * inv);
  }
}

// ---------------- host launcher ----------------
extern "C" void kernel_launch(void* const* d_in, const int* in_sizes, int n_in,
                              void* d_out, int out_size, void* d_ws, size_t ws_size,
                              hipStream_t stream) {
  const float* x    = (const float*)d_in[0];
  const float* wqkv = (const float*)d_in[1];
  const float* wo   = (const float*)d_in[2];
  const float* w1   = (const float*)d_in[3];
  const float* b1   = (const float*)d_in[4];
  const float* w2   = (const float*)d_in[5];
  const float* b2   = (const float*)d_in[6];
  const float* g1   = (const float*)d_in[7];
  const float* be1  = (const float*)d_in[8];
  const float* g2   = (const float*)d_in[9];
  const float* be2  = (const float*)d_in[10];
  float* out = (float*)d_out;

  char* ws = (char*)d_ws;
  size_t off = 0;
  auto alloc = [&](size_t bytes) -> void* {
    void* p = ws + off;
    off += (bytes + 255) & ~(size_t)255;
    return p;
  };
  u16*   wqkvT = (u16*)alloc((size_t)3 * Dd * Dd * 2);       // [3D, D]
  u16*   woT   = (u16*)alloc((size_t)Dd * Dd * 2);           // [D, D]
  u16*   w1T   = (u16*)alloc((size_t)4 * Dd * Dd * 2);       // [4D, D]
  u16*   w2T   = (u16*)alloc((size_t)Dd * 4 * Dd * 2);       // [D, 4D]
  u16*   hln   = (u16*)alloc((size_t)Mm * Dd * 2);
  u16*   qb    = (u16*)alloc((size_t)Mm * Dd * 2);           // [B,H,S,HD]
  u16*   kb    = (u16*)alloc((size_t)Mm * Dd * 2);           // [B,H,S,HD]
  u16*   vTb   = (u16*)alloc((size_t)Mm * Dd * 2);           // [B,H,HD,S]
  u16*   attn  = (u16*)alloc((size_t)Mm * Dd * 2);           // [M, D]
  float* x1    = (float*)alloc((size_t)Mm * Dd * 4);
  u16*   hln2  = (u16*)alloc((size_t)Mm * Dd * 2);
  u16*   gact  = (u16*)alloc((size_t)Mm * 4 * Dd * 2);       // [M, 4D]
  (void)in_sizes; (void)n_in; (void)out_size; (void)ws_size;

  transpose_cast_kernel<<<(3 * Dd * Dd + 255) / 256, 256, 0, stream>>>(wqkv, wqkvT, Dd, 3 * Dd);
  transpose_cast_kernel<<<(Dd * Dd + 255) / 256,     256, 0, stream>>>(wo,   woT,   Dd, Dd);
  transpose_cast_kernel<<<(4 * Dd * Dd + 255) / 256, 256, 0, stream>>>(w1,   w1T,   Dd, 4 * Dd);
  transpose_cast_kernel<<<(4 * Dd * Dd + 255) / 256, 256, 0, stream>>>(w2,   w2T,   4 * Dd, Dd);

  layernorm_kernel<<<Mm, 256, 0, stream>>>(x, g1, be1, hln, Dd);

  gemm_kernel<EPI_QKV><<<dim3(3 * Dd / 128, Mm / 128), 256, 0, stream>>>(
      hln, wqkvT, Mm, 3 * Dd, Dd, nullptr, nullptr, nullptr, nullptr, qb, kb, vTb);

  attn_kernel<<<dim3(Bb * Hh, Ss / 16 / 4), 128, 0, stream>>>(qb, kb, vTb, attn);

  gemm_kernel<EPI_RESID><<<dim3(Dd / 128, Mm / 128), 256, 0, stream>>>(
      attn, woT, Mm, Dd, Dd, nullptr, x, x1, nullptr, nullptr, nullptr, nullptr);

  layernorm_kernel<<<Mm, 256, 0, stream>>>(x1, g2, be2, hln2, Dd);

  gemm_kernel<EPI_BIAS_GELU><<<dim3(4 * Dd / 128, Mm / 128), 256, 0, stream>>>(
      hln2, w1T, Mm, 4 * Dd, Dd, b1, nullptr, nullptr, gact, nullptr, nullptr, nullptr);

  gemm_kernel<EPI_BIAS_RESID><<<dim3(Dd / 128, Mm / 128), 256, 0, stream>>>(
      gact, w2T, Mm, Dd, 4 * Dd, b2, x1, out, nullptr, nullptr, nullptr, nullptr);
}
